// MultiHeadAttentionBlock_36034775614181
// MI455X (gfx1250) — matrix operand order repaired and checked by reading, not compiled, not run on hardware
//
#include <hip/hip_runtime.h>

typedef __bf16 bf16_t;
typedef __attribute__((ext_vector_type(16))) __bf16        v16bf;
typedef __attribute__((ext_vector_type(8)))  float         v8f;
typedef __attribute__((ext_vector_type(4)))  unsigned int  v4u;
typedef __attribute__((ext_vector_type(8)))  int           v8i;
typedef __attribute__((ext_vector_type(4)))  int           v4i;

#define DMODEL 1024
#define NH     16
#define DK     64
#define SEQ    1024
#define NBATCH 2
#define MTOT   (NBATCH * SEQ)
#define NEGINF (-3.4028234663852886e38f)

#if defined(__has_builtin)
#if __has_builtin(__builtin_amdgcn_tensor_load_to_lds) && \
    __has_builtin(__builtin_amdgcn_s_wait_tensorcnt)
#define HAVE_TDM 1
#endif
#endif

// ---------------------------------------------------------------------------
// CDNA5 WMMA bf16 fragment index helpers (wave32 layouts, ISA 7.12.2):
//  A (16x32, MxK): lane r = lane&15 holds row M=r; hf = lane>>4;
//     VGPR i packs K = ((i<4)? 2i : 16+2(i-4)) + 8*hf  and K+1
//  B (32x16, KxN): column N = lane&15; VGPR i packs K = ((i<4)? 2i : 16+2(i-4)) + 8*hf  and K+1
//  C/D (16x16 f32): VGPR i holds M = i + 8*hf, N = lane&15
// ---------------------------------------------------------------------------
__device__ __forceinline__ int a_koff(int i, int hf) {
    return ((i < 4) ? 2 * i : 16 + 2 * (i - 4)) + 8 * hf;
}
__device__ __forceinline__ int b_koff(int i, int hf) {
    return ((i < 4) ? 2 * i : 16 + 2 * (i - 4)) + 8 * hf;
}

#ifdef HAVE_TDM
// ---------------------------------------------------------------------------
// TDM: async 2D bf16 tile load  global[rows x cols, row-major, stride=cols]
// -> LDS, per CDNA5 ISA ch.8 D# layout. One instruction per call, tracked by
// TENSORcnt. Issued by a single wave; EXEC is ignored by tensor ops.
// ---------------------------------------------------------------------------
__device__ __forceinline__ void tdm_load_2d_bf16(unsigned int lds_byte_off,
                                                 const void* gptr,
                                                 unsigned int rows,
                                                 unsigned int cols,
                                                 unsigned int tensor_rows)
{
    const unsigned long long ga = (unsigned long long)gptr;
    v4u g0;
    g0[0] = 1u;                                            // count=1, user desc
    g0[1] = lds_byte_off;                                  // lds_addr
    g0[2] = (unsigned int)(ga & 0xffffffffu);              // global_addr[31:0]
    g0[3] = (unsigned int)((ga >> 32) & 0x01ffffffu)       // global_addr[56:32]
          | (2u << 30);                                    // type = 2 (image)
    v8i g1;
    g1[0] = (int)(1u << 16);                               // data_size=1 -> 2B
    g1[1] = (int)((cols & 0xffffu) << 16);                 // tensor_dim0 lo16
    g1[2] = (int)((tensor_rows & 0xffffu) << 16);          // dim0 hi=0|dim1 lo16
    g1[3] = (int)((cols & 0xffffu) << 16);                 // dim1 hi=0|tile_dim0
    g1[4] = (int)(rows & 0xffffu);                         // tile_dim1 (tile_dim2=0)
    g1[5] = (int)cols;                                     // tensor_dim0_stride lo32
    g1[6] = 0;                                             // stride0 hi | stride1 lo
    g1[7] = 0;
    const v4i z4 = {0, 0, 0, 0};
#if __clang_major__ >= 23
    const v8i z8 = {0, 0, 0, 0, 0, 0, 0, 0};
    __builtin_amdgcn_tensor_load_to_lds(g0, g1, z4, z4, z8, 0);
#else
    __builtin_amdgcn_tensor_load_to_lds(g0, g1, z4, z4, 0);
#endif
}
#endif // HAVE_TDM

// ===========================================================================
// Kernel 1: fused Q/K/V projections.  y = x @ W^T  (W is [out,in], row-major)
//   grid = (N/64, M/16, 3), block = 32: one wave -> 16x64 strip (A reuse x4)
//   Q,K stored [b,h,l,d] bf16 ; V stored transposed [b,h,d,l] bf16
// ===========================================================================
__global__ __launch_bounds__(32)
void qkv_proj_kernel(const float* __restrict__ qx, const float* __restrict__ kx,
                     const float* __restrict__ vx,
                     const float* __restrict__ wq, const float* __restrict__ wk,
                     const float* __restrict__ wv,
                     bf16_t* __restrict__ Qh, bf16_t* __restrict__ Kh,
                     bf16_t* __restrict__ Vt)
{
    const int lane = threadIdx.x & 31;
    const int hf   = lane >> 4;
    const int r    = lane & 15;
    const int n0   = blockIdx.x * 64;
    const int m0   = blockIdx.y * 16;
    const int proj = blockIdx.z;

    const float* x = (proj == 0) ? qx : (proj == 1) ? kx : vx;
    const float* w = (proj == 0) ? wq : (proj == 1) ? wk : wv;

    v8f acc[4] = {};
    for (int k0 = 0; k0 < DMODEL; k0 += 32) {
        const float* xr = x + (size_t)(m0 + r) * DMODEL + k0;   // A row M=r
        __builtin_prefetch(xr + 32, 0, 0);
        v16bf a;
#pragma unroll
        for (int i = 0; i < 8; ++i) {
            const int ka = a_koff(i, hf);
            a[2 * i]     = (bf16_t)xr[ka];
            a[2 * i + 1] = (bf16_t)xr[ka + 1];
        }
#pragma unroll
        for (int nt = 0; nt < 4; ++nt) {
            const float* wn = w + (size_t)(n0 + nt * 16 + r) * DMODEL + k0;
            v16bf b;
#pragma unroll
            for (int i = 0; i < 8; ++i) {
                const int kb = b_koff(i, hf);
                b[2 * i]     = (bf16_t)wn[kb];
                b[2 * i + 1] = (bf16_t)wn[kb + 1];
            }
            acc[nt] = __builtin_amdgcn_wmma_f32_16x16x32_bf16(
                false, a, false, b, (short)0, acc[nt], false, false);
        }
    }
#pragma unroll
    for (int nt = 0; nt < 4; ++nt) {
#pragma unroll
        for (int i = 0; i < 8; ++i) {
            const int m  = m0 + i + 8 * hf;
            const int n  = n0 + nt * 16 + r;
            const int bb = m >> 10;
            const int l  = m & (SEQ - 1);
            const int h  = n >> 6;
            const int d  = n & (DK - 1);
            const bf16_t val = (bf16_t)acc[nt][i];
            if (proj == 0)
                Qh[(((size_t)bb * NH + h) * SEQ + l) * DK + d] = val;
            else if (proj == 1)
                Kh[(((size_t)bb * NH + h) * SEQ + l) * DK + d] = val;
            else
                Vt[(((size_t)bb * NH + h) * DK + d) * SEQ + l] = val;
        }
    }
}

// ===========================================================================
// Kernel 2: fused attention for one (b,h) and one 16-row query tile.
//   K panel TDM-staged to LDS (double-buffered 128x64 bf16 chunks) ->
//   scores (WMMA, f32 in LDS) -> exact 1.5-entmax (bitonic sort + scan) ->
//   P@V (WMMA) -> X[b,l,h*64+d] bf16
//   grid = (SEQ/16, B*H), block = 256 (8 waves)
// ===========================================================================
__global__ __launch_bounds__(256)
void attn_entmax_kernel(const bf16_t* __restrict__ Qh,
                        const bf16_t* __restrict__ Kh,
                        const bf16_t* __restrict__ Vt,
                        const int*    __restrict__ mask,
                        bf16_t*       __restrict__ X)
{
    __shared__ float sS[16][SEQ];       // 64 KB: scores -> x -> probs
    __shared__ float sortBuf[SEQ];      // 4 KB : sorted row, then tau values
    __shared__ float scan1[256];
    __shared__ float scan2[256];
    __shared__ float red[256];
    __shared__ float sTau;
    __shared__ int   sSupport;
#ifdef HAVE_TDM
    __shared__ bf16_t sK[2][128 * DK];  // 2 x 16 KB K-panel staging
#endif

    const int tid  = threadIdx.x;
    const int lane = tid & 31;
    const int wave = tid >> 5;
    const int hf   = lane >> 4;
    const int r    = lane & 15;
    const int l0   = blockIdx.x * 16;   // query-row tile base
    const int bh   = blockIdx.y;        // b*NH + h

    const bf16_t* Qb = Qh + (size_t)bh * SEQ * DK;
    const bf16_t* Kb = Kh + (size_t)bh * SEQ * DK;
    const bf16_t* Vb = Vt + (size_t)bh * DK * SEQ;

    // ---- Phase 1: S = (Q K^T)/sqrt(Dk), masked, into LDS -------------------
    v16bf aq[2];
#pragma unroll
    for (int t = 0; t < 2; ++t) {
        const bf16_t* qr = Qb + (size_t)(l0 + r) * DK + t * 32;
#pragma unroll
        for (int i = 0; i < 8; ++i) {
            const int ka = a_koff(i, hf);
            aq[t][2 * i]     = qr[ka];
            aq[t][2 * i + 1] = qr[ka + 1];
        }
    }

    const int NCHUNK = SEQ / 128;       // 8 chunks; 8 col-tiles = 8 waves each
#ifdef HAVE_TDM
    if (wave == 0) {
        tdm_load_2d_bf16((unsigned int)(unsigned long long)&sK[0][0],
                         Kb, 128, DK, SEQ);
    }
#endif
    for (int c = 0; c < NCHUNK; ++c) {
#ifdef HAVE_TDM
        if (wave == 0) {
            if (c + 1 < NCHUNK) {
                tdm_load_2d_bf16(
                    (unsigned int)(unsigned long long)&sK[(c + 1) & 1][0],
                    Kb + (size_t)(c + 1) * 128 * DK, 128, DK, SEQ);
                __builtin_amdgcn_s_wait_tensorcnt(1);   // chunk c landed
            } else {
                __builtin_amdgcn_s_wait_tensorcnt(0);
            }
        }
        __syncthreads();                 // publish chunk c to all waves
#endif
        const int c0 = c * 128 + wave * 16;   // this wave's 16 K-columns
        v8f acc = {};
#pragma unroll
        for (int t = 0; t < 2; ++t) {
#ifdef HAVE_TDM
            const bf16_t* kr = &sK[c & 1][(wave * 16 + r) * DK + t * 32];
#else
            const bf16_t* kr = Kb + (size_t)(c0 + r) * DK + t * 32;
#endif
            v16bf b;
#pragma unroll
            for (int i = 0; i < 8; ++i) {
                const int kb = b_koff(i, hf);
                b[2 * i]     = kr[kb];
                b[2 * i + 1] = kr[kb + 1];
            }
            acc = __builtin_amdgcn_wmma_f32_16x16x32_bf16(
                false, aq[t], false, b, (short)0, acc, false, false);
        }
#pragma unroll
        for (int i = 0; i < 8; ++i) {
            const int row = i + 8 * hf;
            const int col = c0 + r;
            float sc = acc[i] * 0.125f;                 // 1/sqrt(64)
            if (mask[(size_t)(l0 + row) * SEQ + col] == 0) sc = NEGINF;
            sS[row][col] = sc;
        }
#ifdef HAVE_TDM
        __syncthreads();  // all waves done reading sK[c&1] before next TDM lands
#endif
    }
    __syncthreads();

    // ---- Phase 2: exact 1.5-entmax per row (rows sequential, block coop) ---
    for (int row = 0; row < 16; ++row) {
        // row max of x = s/2
        float mx = NEGINF;
        for (int j = tid; j < SEQ; j += 256) mx = fmaxf(mx, sS[row][j] * 0.5f);
        red[tid] = mx;
        __syncthreads();
        for (int off = 128; off > 0; off >>= 1) {
            if (tid < off) red[tid] = fmaxf(red[tid], red[tid + off]);
            __syncthreads();
        }
        const float rowmax = red[0];
        __syncthreads();

        // x = s/2 - max ; keep in sS, copy to sortBuf
        for (int j = tid; j < SEQ; j += 256) {
            const float xv = sS[row][j] * 0.5f - rowmax;
            sS[row][j] = xv;
            sortBuf[j] = xv;
        }
        __syncthreads();

        // bitonic sort, descending
        for (int kk = 2; kk <= SEQ; kk <<= 1) {
            for (int jj = kk >> 1; jj > 0; jj >>= 1) {
                for (int idx = tid; idx < SEQ; idx += 256) {
                    const int partner = idx ^ jj;
                    if (partner > idx) {
                        const float a = sortBuf[idx];
                        const float b = sortBuf[partner];
                        const bool desc = ((idx & kk) == 0);
                        if (desc ? (a < b) : (a > b)) {
                            sortBuf[idx]     = b;
                            sortBuf[partner] = a;
                        }
                    }
                }
                __syncthreads();
            }
        }

        // segmented scan: thread owns 4 consecutive sorted elems
        float xs[4];
        float s1 = 0.0f, s2 = 0.0f;
#pragma unroll
        for (int u = 0; u < 4; ++u) {
            xs[u] = sortBuf[tid * 4 + u];
            s1 += xs[u];
            s2 += xs[u] * xs[u];
        }
        scan1[tid] = s1;
        scan2[tid] = s2;
        __syncthreads();
        for (int off = 1; off < 256; off <<= 1) {
            const float a1 = scan1[tid];
            const float a2 = scan2[tid];
            float b1 = 0.0f, b2 = 0.0f;
            if (tid >= off) { b1 = scan1[tid - off]; b2 = scan2[tid - off]; }
            __syncthreads();
            scan1[tid] = a1 + b1;
            scan2[tid] = a2 + b2;
            __syncthreads();
        }
        float run1 = scan1[tid] - s1;   // exclusive prefix
        float run2 = scan2[tid] - s2;

        int cnt = 0;
        float tl[4];
#pragma unroll
        for (int u = 0; u < 4; ++u) {
            const int e = tid * 4 + u;
            const float xv = xs[u];
            run1 += xv;
            run2 += xv * xv;
            const float rho    = (float)(e + 1);
            const float mean   = run1 / rho;
            const float meansq = run2 / rho;
            const float ssv    = rho * (meansq - mean * mean);
            const float delta  = (1.0f - ssv) / rho;
            const float tau    = mean - sqrtf(fmaxf(delta, 0.0f));
            if (tau <= xv) cnt++;
            tl[u] = tau;
        }
#pragma unroll
        for (int u = 0; u < 4; ++u) sortBuf[tid * 4 + u] = tl[u];  // tau array
        if (tid == 0) sSupport = 0;
        __syncthreads();
        atomicAdd(&sSupport, cnt);
        __syncthreads();
        if (tid == 0) {
            int si = sSupport - 1;
            if (si < 0) si = 0;
            sTau = sortBuf[si];
        }
        __syncthreads();
        const float tau_star = sTau;

        // p = max(x - tau*, 0)^2  back into sS
        for (int j = tid; j < SEQ; j += 256) {
            float pv = sS[row][j] - tau_star;
            pv = fmaxf(pv, 0.0f);
            sS[row][j] = pv * pv;
        }
        __syncthreads();
    }

    // ---- Phase 3: X_tile(16x64) = P(16x1024) @ V(1024x64), waves 0..3 -----
    if (wave < 4) {
        const int n0 = wave * 16;   // output-dim tile within Dk=64
        v8f acc = {};
        for (int k0 = 0; k0 < SEQ; k0 += 32) {
            const float*  pr = &sS[r][k0];                        // A row M=r
            const bf16_t* vr = Vb + (size_t)(n0 + r) * SEQ + k0;  // B col N=r (V^T)
            __builtin_prefetch(vr + 128, 0, 0);
            v16bf a, b;
#pragma unroll
            for (int i = 0; i < 8; ++i) {
                const int ka = a_koff(i, hf);
                a[2 * i]     = (bf16_t)pr[ka];
                a[2 * i + 1] = (bf16_t)pr[ka + 1];
                const int kb = b_koff(i, hf);
                b[2 * i]     = vr[kb];
                b[2 * i + 1] = vr[kb + 1];
            }
            acc = __builtin_amdgcn_wmma_f32_16x16x32_bf16(
                false, a, false, b, (short)0, acc, false, false);
        }
        const int bidx = bh >> 4;       // b
        const int h    = bh & 15;       // head
#pragma unroll
        for (int i = 0; i < 8; ++i) {
            const int l = l0 + i + 8 * hf;
            const int d = n0 + r;
            X[((size_t)bidx * SEQ + l) * DMODEL + h * DK + d] = (bf16_t)acc[i];
        }
    }
}

// ===========================================================================
// Kernel 3: output projection  out = X @ Wo^T   (f32 out)
//   grid = (N/64, M/16), block = 32: one wave -> 16x64 strip (A reuse x4)
// ===========================================================================
__global__ __launch_bounds__(32)
void out_proj_kernel(const bf16_t* __restrict__ X, const float* __restrict__ wo,
                     float* __restrict__ out)
{
    const int lane = threadIdx.x & 31;
    const int hf   = lane >> 4;
    const int r    = lane & 15;
    const int n0   = blockIdx.x * 64;
    const int m0   = blockIdx.y * 16;

    v8f acc[4] = {};
    for (int k0 = 0; k0 < DMODEL; k0 += 32) {
        const bf16_t* xr = X + (size_t)(m0 + r) * DMODEL + k0;
        v16bf a;
#pragma unroll
        for (int i = 0; i < 8; ++i) {
            const int ka = a_koff(i, hf);
            a[2 * i]     = xr[ka];
            a[2 * i + 1] = xr[ka + 1];
        }
#pragma unroll
        for (int nt = 0; nt < 4; ++nt) {
            const float* wn = wo + (size_t)(n0 + nt * 16 + r) * DMODEL + k0;
            __builtin_prefetch(wn + 32, 0, 0);
            v16bf b;
#pragma unroll
            for (int i = 0; i < 8; ++i) {
                const int kb = b_koff(i, hf);
                b[2 * i]     = (bf16_t)wn[kb];
                b[2 * i + 1] = (bf16_t)wn[kb + 1];
            }
            acc[nt] = __builtin_amdgcn_wmma_f32_16x16x32_bf16(
                false, a, false, b, (short)0, acc[nt], false, false);
        }
    }
#pragma unroll
    for (int nt = 0; nt < 4; ++nt) {
#pragma unroll
        for (int i = 0; i < 8; ++i) {
            const int m = m0 + i + 8 * hf;
            const int n = n0 + nt * 16 + r;
            out[(size_t)m * DMODEL + n] = acc[nt][i];
        }
    }
}

// ===========================================================================
extern "C" void kernel_launch(void* const* d_in, const int* in_sizes, int n_in,
                              void* d_out, int out_size, void* d_ws, size_t ws_size,
                              hipStream_t stream)
{
    (void)in_sizes; (void)n_in; (void)out_size; (void)ws_size;

    const float* q    = (const float*)d_in[0];
    const float* k    = (const float*)d_in[1];
    const float* v    = (const float*)d_in[2];
    const int*   mask = (const int*)  d_in[3];
    const float* wq   = (const float*)d_in[4];
    const float* wk   = (const float*)d_in[5];
    const float* wv   = (const float*)d_in[6];
    const float* wo   = (const float*)d_in[7];
    float*       out  = (float*)d_out;

    // workspace: 4 bf16 arrays of B*H*L*Dk (=B*L*D) elements = 4 MB each
    const size_t ARR = (size_t)NBATCH * NH * SEQ * DK * sizeof(bf16_t);
    char* wsb = (char*)d_ws;
    bf16_t* Qh = (bf16_t*)(wsb + 0 * ARR);
    bf16_t* Kh = (bf16_t*)(wsb + 1 * ARR);
    bf16_t* Vt = (bf16_t*)(wsb + 2 * ARR);   // V transposed [b,h,d,l]
    bf16_t* X  = (bf16_t*)(wsb + 3 * ARR);   // [b,l,h*Dk]

    qkv_proj_kernel<<<dim3(DMODEL / 64, MTOT / 16, 3), 32, 0, stream>>>(
        q, k, v, wq, wk, wv, Qh, Kh, Vt);

    attn_entmax_kernel<<<dim3(SEQ / 16, NBATCH * NH), 256, 0, stream>>>(
        Qh, Kh, Vt, mask, X);

    out_proj_kernel<<<dim3(DMODEL / 64, MTOT / 16), 32, 0, stream>>>(X, wo, out);
}
